// Spectral_146028888508
// MI455X (gfx1250) — matrix operand order repaired and checked by reading, not compiled, not run on hardware
//
#include <hip/hip_runtime.h>
#include <stdint.h>
#include <stddef.h>

// ---------------------------------------------------------------------------
// MI455X (gfx1250) fused Spectral kernel.
// Roofline: ~84 MB unavoidable HBM traffic (spec in + out) @23.3TB/s ~ 3.6us;
// ~20 GFLOP of matmul -> run on v_wmma_f32_16x16x32_bf16 (f32 accum) to stay
// memory-bound. Weights (<600KB) pre-packed to bf16 WMMA-B layout in d_ws;
// they live in the 192MB L2 so per-tile rereads are L2 hits.
// One wave32 per 16-row batch tile; all activations in LDS; no barriers
// (single-wave workgroup => wave-lockstep, per-wave-ordered LDS).
// v3: A-operand tiles hoisted to register arrays (loaded/converted once per
// cell, not once per gate tile); output tile staged in LDS (reusing the dead
// SPEC buffer) and flushed with coalesced global_store_b128.
// ---------------------------------------------------------------------------

typedef __attribute__((ext_vector_type(8)))  float  v8f;
typedef __attribute__((ext_vector_type(16))) __bf16 v16bf;
typedef __attribute__((ext_vector_type(2)))  __bf16 bf2;

#define WMMA_BF16(a, b, c) \
  __builtin_amdgcn_wmma_f32_16x16x32_bf16(false, (a), false, (b), (short)0, (c), false, false)

__device__ __forceinline__ float sigm(float x) { return 1.0f / (1.0f + __expf(-x)); }

// --- A operand: 16x32 f32 tile in LDS -> bf16 A layout ---------------------
// A layout matching packed B: lane m=lane&15, hi=lane>>4,
// VGPR j holds K = hi*16 + 2*j (+1).
// KLIM==0: no masking (K tail known zero-padded / full chunk).
// KLIM>0:  unconditional load + value select (reads stay inside LDS arena).
template <int KLIM>
__device__ __forceinline__ v16bf load_A(const float* A, int stride, int kc) {
  const int lane = threadIdx.x;
  const int m = lane & 15, hi = lane >> 4;
  union { v16bf v; unsigned int u[8]; } a;
#pragma unroll
  for (int j = 0; j < 8; ++j) {
    const int k = kc + hi * 16 + (j << 1);
    float f0 = A[m * stride + k];
    float f1 = A[m * stride + k + 1];
    if (KLIM > 0) {                 // folds per-j (and per-lane-half) at compile time
      f0 = (k     < KLIM) ? f0 : 0.0f;
      f1 = (k + 1 < KLIM) ? f1 : 0.0f;
    }
    bf2 p;
    p[0] = (__bf16)f0;
    p[1] = (__bf16)f1;
    a.u[j] = __builtin_bit_cast(unsigned int, p);
  }
  return a.v;
}

// --- B operand: prepacked chunk (32 lanes x 16 bf16). lane holds column
// n=lane&15, K = (lane>>4)*16 + e (contiguous K order, same as load_A). -------------
__device__ __forceinline__ v16bf load_B(const unsigned short* chunk) {
  return *reinterpret_cast<const v16bf*>(chunk + threadIdx.x * 16);
}

// --- conv3(+stride) -> softmax ("trans" block), rows in lanes --------------
__device__ __forceinline__ void trans_op(const float* src, int ss, int Lin, int stride,
                                         float* dst, int ds, int dcol, const float* w) {
  const int lane = threadIdx.x;
  const float w0 = w[0], w1 = w[1], w2 = w[2];
  const int Lout = Lin / stride;
  const int total = 16 * Lout;
  for (int t = lane; t < total; t += 32) {
    int r = t / Lout, oc = t - r * Lout;
    int ic = oc * stride;
    float acc = w1 * src[r * ss + ic];
    if (ic > 0)       acc += w0 * src[r * ss + ic - 1];
    if (ic + 1 < Lin) acc += w2 * src[r * ss + ic + 1];
    dst[r * ds + dcol + oc] = acc;
  }
  if (lane < 16) {  // serial per-row softmax (tiny)
    float mx = -3.4e38f;
    for (int i = 0; i < Lout; ++i) mx = fmaxf(mx, dst[lane * ds + dcol + i]);
    float s = 0.0f;
    for (int i = 0; i < Lout; ++i) {
      float e = __expf(dst[lane * ds + dcol + i] - mx);
      dst[lane * ds + dcol + i] = e;
      s += e;
    }
    float r = 1.0f / s;
    for (int i = 0; i < Lout; ++i) dst[lane * ds + dcol + i] *= r;
  }
}

// --- LSTM cell, HDIM >= 16. A tiles hoisted to registers; all 4 gate tiles
// accumulated in registers so z never hits LDS. Accum element: m=j+8*hi,
// n=lane&15 (ISA C/D layout). ------------------------------------------------
template <int DIN, int HDIM, bool HAS_H>
__device__ __forceinline__ void cell16(const float* xin, int sx,
                                       const float* hin, const float* cin, int sp,
                                       float* hout, float* cout, int so, float* ghout,
                                       const unsigned short* Wih, const unsigned short* Whh,
                                       const float* bsum) {
  constexpr int GT  = HDIM / 16;      // n-tiles per gate
  constexpr int NT  = 4 * GT;         // n-tiles in z
  constexpr int KCx = (DIN  + 31) / 32;
  constexpr int KCh = (HDIM + 31) / 32;
  constexpr int MX  = (DIN  % 32 == 0) ? 0 : DIN;   // X slices here are 32-mult
  constexpr int MH  = (HDIM % 32 == 0) ? 0 : HDIM;  // mask h when HDIM==16
  const int lane = threadIdx.x, n = lane & 15, hi = lane >> 4;

  v16bf ax[KCx];
#pragma unroll
  for (int kc = 0; kc < KCx; ++kc) ax[kc] = load_A<MX>(xin, sx, kc * 32);
  v16bf ah[KCh];
  if constexpr (HAS_H) {
#pragma unroll
    for (int kc = 0; kc < KCh; ++kc) ah[kc] = load_A<MH>(hin, sp, kc * 32);
  }

#pragma unroll
  for (int nt = 0; nt < GT; ++nt) {
    v8f ai = {0,0,0,0,0,0,0,0}, af = ai, ag = ai, ao = ai;
#pragma unroll
    for (int kc = 0; kc < KCx; ++kc) {
      const unsigned short* b0 = Wih + (size_t)(kc * NT) * 512;
      ai = WMMA_BF16(ax[kc], load_B(b0 + (size_t)(nt         ) * 512), ai);
      af = WMMA_BF16(ax[kc], load_B(b0 + (size_t)(nt +     GT) * 512), af);
      ag = WMMA_BF16(ax[kc], load_B(b0 + (size_t)(nt + 2 * GT) * 512), ag);
      ao = WMMA_BF16(ax[kc], load_B(b0 + (size_t)(nt + 3 * GT) * 512), ao);
    }
    if constexpr (HAS_H) {
#pragma unroll
      for (int kc = 0; kc < KCh; ++kc) {
        const unsigned short* b0 = Whh + (size_t)(kc * NT) * 512;
        ai = WMMA_BF16(ah[kc], load_B(b0 + (size_t)(nt         ) * 512), ai);
        af = WMMA_BF16(ah[kc], load_B(b0 + (size_t)(nt +     GT) * 512), af);
        ag = WMMA_BF16(ah[kc], load_B(b0 + (size_t)(nt + 2 * GT) * 512), ag);
        ao = WMMA_BF16(ah[kc], load_B(b0 + (size_t)(nt + 3 * GT) * 512), ao);
      }
    }
    const float bi = bsum[0 * HDIM + nt * 16 + n];
    const float bf = bsum[1 * HDIM + nt * 16 + n];
    const float bg = bsum[2 * HDIM + nt * 16 + n];
    const float bo = bsum[3 * HDIM + nt * 16 + n];
#pragma unroll
    for (int j = 0; j < 8; ++j) {
      const int m = j + 8 * hi;
      float iv = sigm(ai[j] + bi);
      float fv = sigm(af[j] + bf);
      float gv = tanhf(ag[j] + bg);
      float ov = sigm(ao[j] + bo);
      float cp = HAS_H ? cin[m * sp + nt * 16 + n] : 0.0f;
      float cv = fv * cp + iv * gv;
      float hv = ov * tanhf(cv);
      if (cout)  cout[m * so + nt * 16 + n] = cv;
      if (hout)  hout[m * so + nt * 16 + n] = hv;
      if (ghout) ghout[m * 128 + nt * 16 + n] = hv;   // LDS OUT tile, stride 128
    }
  }
}

// --- LSTM cell, HDIM == 8: z is 16x32 = 2 tiles: [i|f] and [g|o]. Pair the
// gates across lanes with __shfl_xor(.,8); lanes n<8 own output column n. ----
template <int DIN, bool HAS_H>
__device__ __forceinline__ void cell8(const float* xin, int sx,
                                      const float* hin, const float* cin, int sp,
                                      float* hout, float* cout, int so, float* gcout,
                                      const unsigned short* Wih, const unsigned short* Whh,
                                      const float* bsum) {
  constexpr int KCx = (DIN + 31) / 32;
  constexpr int MX  = (DIN % 32 == 0) ? 0 : DIN;  // DIN==16 -> masked
  const int lane = threadIdx.x, n = lane & 15, hi = lane >> 4;
  v8f a0 = {0,0,0,0,0,0,0,0}, a1 = a0;
#pragma unroll
  for (int kc = 0; kc < KCx; ++kc) {
    v16bf a = load_A<MX>(xin, sx, kc * 32);
    const unsigned short* b0 = Wih + (size_t)(kc * 2) * 512;
    a0 = WMMA_BF16(a, load_B(b0      ), a0);
    a1 = WMMA_BF16(a, load_B(b0 + 512), a1);
  }
  if constexpr (HAS_H) {
    v16bf a = load_A<8>(hin, sp, 0);
    a0 = WMMA_BF16(a, load_B(Whh      ), a0);
    a1 = WMMA_BF16(a, load_B(Whh + 512), a1);
  }
  const float b0 = bsum[n];        // i (n<8) / f (n>=8)
  const float b1 = bsum[16 + n];   // g (n<8) / o (n>=8)
#pragma unroll
  for (int j = 0; j < 8; ++j) {
    float v0 = a0[j] + b0;
    float v1 = a1[j] + b1;
    float fv = __shfl_xor(v0, 8, 32);  // all lanes participate
    float ov = __shfl_xor(v1, 8, 32);
    if (n < 8) {
      const int m = j + 8 * hi;
      float iv = sigm(v0), gv = tanhf(v1);
      fv = sigm(fv); ov = sigm(ov);
      float cp = HAS_H ? cin[m * sp + n] : 0.0f;
      float cv = fv * cp + iv * gv;
      if (cout)  cout[m * so + n] = cv;
      if (gcout) gcout[m * 128 + n] = cv;  // LDS OUT tile, stride 128
      if (hout)  hout[m * so + n] = ov * tanhf(cv);
    }
  }
}

// --------------------------- kernel parameters -----------------------------
struct KParams {
  const float* spec;
  float* out;
  const float* pre_b;
  const unsigned short* preW;
  const unsigned short* Wih[2][4];   // [0]=fwd [1]=bwd, cells c1..c4
  const unsigned short* Whh[2][4];
  const float* bsum[2][4];
  const float* convk[12];  // 0:pre; 1..5: t12h,t12c,t23h,t23c,t34h; 6..11: t43h,t43c,t32h,t32c,t21h,t21c
  int rows;
};

// ------------------------------ main kernel --------------------------------
// LDS arena (floats): X 16x130=2080, 6 small 16x68=1088, SPEC 16x228=3648.
// SPEC is dead after the pre-GEMM; its first 2048 floats are reused as the
// 16x128 OUT staging tile (flushed with coalesced b128 stores).
__global__ __launch_bounds__(32) void spectral_wmma_kernel(KParams P) {
  __shared__ float lds[2080 + 6 * 1088 + 3648];   // 49,024 bytes
  float* X    = lds;
  float* H    = lds + 2080;
  float* C    = H   + 1088;
  float* AH   = C   + 1088;
  float* AC   = AH  + 1088;
  float* BH   = AC  + 1088;
  float* BC   = BH  + 1088;
  float* SPEC = BC  + 1088;
  float* OUT  = SPEC;              // alias: OUT tile (16 x 128) after pre-GEMM
  const int SX = 130, SB = 68, SS = 228;
  const int lane = threadIdx.x, n = lane & 15, hi = lane >> 4;
  const int row0 = blockIdx.x << 4;
  if (row0 >= P.rows) return;

  // ---- stage spec tile -> LDS (coalesced b128), zero-pad K 200 -> 228 -----
  {
    const float4* src = reinterpret_cast<const float4*>(P.spec + (size_t)row0 * 200);
    for (int t = lane; t < 16 * 50; t += 32) {       // 16 rows x 50 float4
      int r = t / 50, q = t - r * 50;
      float4 v = src[t];
      *reinterpret_cast<float4*>(SPEC + r * SS + q * 4) = v;
    }
    for (int t = lane; t < 16 * 28; t += 32) {       // pad cols 200..227
      int r = t / 28, q = t - r * 28;
      SPEC[r * SS + 200 + q] = 0.0f;
    }
  }

  // ---- pre: X = spec @ preW^T + pre_b  (K padded to 224, zeros in LDS) ----
  {
    v16bf as[7];
#pragma unroll
    for (int kc = 0; kc < 7; ++kc) as[kc] = load_A<0>(SPEC, SS, kc * 32);
#pragma unroll
    for (int nt = 0; nt < 8; ++nt) {
      v8f acc = {0,0,0,0,0,0,0,0};
#pragma unroll
      for (int kc = 0; kc < 7; ++kc)
        acc = WMMA_BF16(as[kc], load_B(P.preW + ((size_t)(kc * 8 + nt) << 9)), acc);
      float bias = P.pre_b[nt * 16 + n];
#pragma unroll
      for (int j = 0; j < 8; ++j)
        X[(j + 8 * hi) * SX + nt * 16 + n] = acc[j] + bias;
    }
  }
  // ---- pre_conv (k=3 pad=1) + tanh, in place, one row per lane ------------
  {
    const float w0 = P.convk[0][0], w1 = P.convk[0][1], w2 = P.convk[0][2];
    if (lane < 16) {
      float prev = 0.0f, cur = X[lane * SX];
      for (int i = 0; i < 128; ++i) {
        float nxt = (i < 127) ? X[lane * SX + i + 1] : 0.0f;
        X[lane * SX + i] = tanhf(w0 * prev + w1 * cur + w2 * nxt);
        prev = cur; cur = nxt;
      }
    }
  }

  // =========================== forward pyramid ============================
  cell16<128, 64, false>(X, SX, nullptr, nullptr, SB, H, C, SB, nullptr,
                         P.Wih[0][0], P.Whh[0][0], P.bsum[0][0]);
  trans_op(H, SB, 64, 2, AH, SB, 0, P.convk[1]);   // t12h
  trans_op(C, SB, 64, 2, AC, SB, 0, P.convk[2]);   // t12c
  for (int i = 0; i < 2; ++i) {
    cell16<64, 32, true>(X + i * 64, SX, AH, AC, SB, H, C, SB, nullptr,
                         P.Wih[0][1], P.Whh[0][1], P.bsum[0][1]);
    trans_op(H, SB, 32, 2, BH, SB, i * 16, P.convk[3]);  // t23h
    trans_op(C, SB, 32, 2, BC, SB, i * 16, P.convk[4]);  // t23c
  }
  for (int i = 0; i < 4; ++i) {
    cell16<32, 16, true>(X + i * 32, SX, BH + (i >> 1) * 16, BC + (i >> 1) * 16, SB,
                         H, C, SB, nullptr, P.Wih[0][2], P.Whh[0][2], P.bsum[0][2]);
    trans_op(H, SB, 16, 2, AH, SB, i * 8, P.convk[5]);   // t34h
    trans_op(C, SB, 16, 2, AC, SB, i * 8, P.convk[5]);   // t34h reused (faithful to source)
  }
  for (int i = 0; i < 8; ++i)  // level-4 cell states -> OUT[:,0:64] (LDS)
    cell8<16, true>(X + i * 16, SX, AH + (i >> 1) * 8, AC + (i >> 1) * 8, SB,
                    nullptr, nullptr, SB, OUT + i * 8,
                    P.Wih[0][3], P.Whh[0][3], P.bsum[0][3]);

  // =========================== backward pyramid ===========================
  for (int i = 0; i < 8; ++i) {  // zero state each step (faithful to source)
    cell8<16, false>(X + i * 16, SX, nullptr, nullptr, SB,
                     H + (i & 1) * 8, C + (i & 1) * 8, SB, nullptr,
                     P.Wih[1][3], P.Whh[1][3], P.bsum[1][3]);
    if (i & 1) {
      trans_op(H, SB, 16, 1, AH, SB, (i >> 1) * 16, P.convk[6]);  // t43h
      trans_op(C, SB, 16, 1, AC, SB, (i >> 1) * 16, P.convk[7]);  // t43c
    }
  }
  for (int i = 0; i < 4; ++i) {
    cell16<32, 16, true>(X + i * 32, SX, AH + i * 16, AC + i * 16, SB,
                         H + (i & 1) * 16, C + (i & 1) * 16, SB, nullptr,
                         P.Wih[1][2], P.Whh[1][2], P.bsum[1][2]);
    if (i & 1) {
      trans_op(H, SB, 32, 1, BH, SB, (i >> 1) * 32, P.convk[8]);  // t32h
      trans_op(C, SB, 32, 1, BC, SB, (i >> 1) * 32, P.convk[9]);  // t32c
    }
  }
  for (int i = 0; i < 2; ++i) {
    cell16<64, 32, true>(X + i * 64, SX, BH + i * 32, BC + i * 32, SB,
                         H + (i & 1) * 32, C + (i & 1) * 32, SB, nullptr,
                         P.Wih[1][1], P.Whh[1][1], P.bsum[1][1]);
    if (i & 1) {
      trans_op(H, SB, 64, 1, AH, SB, 0, P.convk[10]);  // t21h
      trans_op(C, SB, 64, 1, AC, SB, 0, P.convk[11]);  // t21c
    }
  }
  cell16<128, 64, true>(X, SX, AH, AC, SB, nullptr, nullptr, SB, OUT + 64,
                        P.Wih[1][0], P.Whh[1][0], P.bsum[1][0]);  // h1 -> OUT[:,64:128]

  // ---- flush OUT tile: 512 coalesced b128 stores (contiguous 8KB) ---------
  {
    float4* dst = reinterpret_cast<float4*>(P.out + (size_t)row0 * 128);
    const float4* srcv = reinterpret_cast<const float4*>(OUT);
    for (int t = lane; t < 16 * 32; t += 32)
      dst[t] = srcv[t];
  }
}

// ----------------------- weight / bias prep kernels ------------------------
__device__ __forceinline__ unsigned short f2bf(float v) {  // RNE f32->bf16
  unsigned int x = __float_as_uint(v);
  unsigned int r = x + 0x7FFFu + ((x >> 16) & 1u);
  if ((x & 0x7F800000u) == 0x7F800000u) r = x;  // inf/nan passthrough
  return (unsigned short)(r >> 16);
}

// Pack W (N,K) row-major into per-(kc,nt) 32x16 bf16 chunks:
// chunk = kc*NT + nt; within chunk element (lane2, e): n = nt*16 + (lane2&15),
// k = kc*32 + (lane2>>4)*16 + e. Zero pad out of range.
__global__ void pack_weight(const float* __restrict__ W, unsigned short* __restrict__ dst,
                            int N, int K) {
  const int NT = (N + 15) >> 4;
  const int KC = (K + 31) >> 5;
  const int total = NT * KC * 512;
  for (int t = blockIdx.x * blockDim.x + threadIdx.x; t < total; t += gridDim.x * blockDim.x) {
    int chunk  = t >> 9;
    int within = t & 511;
    int lane2  = within >> 4;
    int e      = within & 15;
    int kc = chunk / NT, nt = chunk - kc * NT;
    int ng = nt * 16 + (lane2 & 15);
    int k  = kc * 32 + (lane2 >> 4) * 16 + e;
    float v = (ng < N && k < K) ? W[(size_t)ng * K + k] : 0.0f;
    dst[t] = f2bf(v);
  }
}

__global__ void pack_bias(const float* __restrict__ bih, const float* __restrict__ bhh,
                          float* __restrict__ dst, int nel) {
  int t = blockIdx.x * blockDim.x + threadIdx.x;
  if (t < nel) dst[t] = bih[t] + bhh[t];
}

// -------------------------------- host side --------------------------------
extern "C" void kernel_launch(void* const* d_in, const int* in_sizes, int n_in,
                              void* d_out, int out_size, void* d_ws, size_t ws_size,
                              hipStream_t stream) {
  (void)n_in; (void)out_size; (void)ws_size;
  // Resolve input flattening: insertion-order (spec first) vs jax tree_leaves
  // sorted-key order (bwd..., fwd..., pre_W, pre_b, pre_conv_w, spec).
  const bool insertion = in_sizes[0] > 1000000;  // spec = 65536*200
  int I_spec, I_preW, I_preB;
  int I_Wih[2][4], I_Whh[2][4], I_bih[2][4], I_bhh[2][4];
  int I_conv[12];  // kernel convk mapping
  if (insertion) {
    I_spec = 0; I_preW = 1; I_preB = 2; I_conv[0] = 3;
    int base = 4;
    for (int c = 0; c < 4; ++c) { I_Wih[0][c] = base; I_Whh[0][c] = base + 1;
                                  I_bih[0][c] = base + 2; I_bhh[0][c] = base + 3; base += 4; }
    I_conv[1] = 20; I_conv[2] = 21; I_conv[3] = 22; I_conv[4] = 23; I_conv[5] = 24;  // fwd t
    base = 26;
    for (int c = 0; c < 4; ++c) { I_Wih[1][c] = base; I_Whh[1][c] = base + 1;
                                  I_bih[1][c] = base + 2; I_bhh[1][c] = base + 3; base += 4; }
    I_conv[6] = 42; I_conv[7] = 43; I_conv[8] = 44; I_conv[9] = 45; I_conv[10] = 46; I_conv[11] = 47;
  } else {
    for (int c = 0; c < 4; ++c) { I_Whh[1][c] = 4 * c; I_Wih[1][c] = 4 * c + 1;
                                  I_bhh[1][c] = 4 * c + 2; I_bih[1][c] = 4 * c + 3; }
    // 16..21 = t21c,t21h,t32c,t32h,t43c,t43h
    I_conv[6] = 21; I_conv[7] = 20; I_conv[8] = 19; I_conv[9] = 18; I_conv[10] = 17; I_conv[11] = 16;
    for (int c = 0; c < 4; ++c) { I_Whh[0][c] = 22 + 4 * c; I_Wih[0][c] = 23 + 4 * c;
                                  I_bhh[0][c] = 24 + 4 * c; I_bih[0][c] = 25 + 4 * c; }
    // 38..43 = t12c,t12h,t23c,t23h,t34c,t34h
    I_conv[1] = 39; I_conv[2] = 38; I_conv[3] = 41; I_conv[4] = 40; I_conv[5] = 43;
    I_preW = 44; I_preB = 45; I_conv[0] = 46; I_spec = 47;
  }
  auto F = [&](int i) -> const float* { return (const float*)d_in[i]; };

  // Workspace sub-allocation (256B aligned).
  size_t off = 0;
  auto alloc = [&](size_t bytes) -> char* {
    off = (off + 255) & ~(size_t)255;
    char* p = (char*)d_ws + off;
    off += bytes;
    return p;
  };
  auto pack = [&](const float* W, int N, int K) -> const unsigned short* {
    int NT = (N + 15) / 16, KC = (K + 31) / 32;
    size_t el = (size_t)NT * KC * 512;
    unsigned short* dst = (unsigned short*)alloc(el * 2);
    int blocks = (int)((el + 255) / 256);
    pack_weight<<<blocks, 256, 0, stream>>>(W, dst, N, K);
    return dst;
  };
  auto packb = [&](const float* bih, const float* bhh, int nel) -> const float* {
    float* dst = (float*)alloc((size_t)nel * 4);
    pack_bias<<<(nel + 127) / 128, 128, 0, stream>>>(bih, bhh, dst, nel);
    return dst;
  };

  static const int hdims[4] = {64, 32, 16, 8};
  static const int dins[4]  = {128, 64, 32, 16};

  KParams P;
  P.spec  = F(I_spec);
  P.out   = (float*)d_out;
  P.pre_b = F(I_preB);
  P.preW  = pack(F(I_preW), 128, 200);
  for (int d = 0; d < 2; ++d)
    for (int c = 0; c < 4; ++c) {
      P.Wih[d][c]  = pack(F(I_Wih[d][c]), 4 * hdims[c], dins[c]);
      P.Whh[d][c]  = pack(F(I_Whh[d][c]), 4 * hdims[c], hdims[c]);
      P.bsum[d][c] = packb(F(I_bih[d][c]), F(I_bhh[d][c]), 4 * hdims[c]);
    }
  for (int i = 0; i < 12; ++i) P.convk[i] = F(I_conv[i]);
  P.rows = in_sizes[I_spec] / 200;

  const int tiles = (P.rows + 15) / 16;  // 4096 single-wave workgroups
  spectral_wmma_kernel<<<tiles, 32, 0, stream>>>(P);
}
